// TreeSoftDiceLoss_18305150616185
// MI455X (gfx1250) — 8-bit matrix operand order repaired and checked by reading, compiled, not run on hardware
//
#include <hip/hip_runtime.h>
#include <stdint.h>


typedef __attribute__((ext_vector_type(8))) int v8i;

__global__ void tsd_init_counters(unsigned* __restrict__ g) {
    if (threadIdx.x < 3) g[threadIdx.x] = 0u;
}

__global__ __launch_bounds__(256) void tsd_class_hist_wmma(
        const uint4* __restrict__ t4, long long n4,
        const int*  __restrict__ t,  long long n,
        unsigned*   __restrict__ gcnt) {
    const long long tid    = (long long)blockIdx.x * blockDim.x + threadIdx.x;
    const long long stride = (long long)gridDim.x * blockDim.x;

    unsigned acc = 0u;
    for (long long i = tid; i < n4; i += stride) {
        if (i + 4 * stride < n4)
            __builtin_prefetch(&t4[i + 4 * stride], 0, 0);
        uint4 v = t4[i];
        acc += 1u << ((v.x & 3u) << 3);
        acc += 1u << ((v.y & 3u) << 3);
        acc += 1u << ((v.z & 3u) << 3);
        acc += 1u << ((v.w & 3u) << 3);
    }
    for (long long i = (n4 << 2) + tid; i < n; i += stride)
        acc += 1u << (((unsigned)t[i] & 3u) << 3);

    unsigned lane = threadIdx.x & 31u;
    v8i A = {0, 0, 0, 0, 0, 0, 0, 0};
    A[0] = (int)acc;
    unsigned sel = ((lane & 15u) < 3u) ? (1u << ((lane & 15u) << 3)) : 0u;
    v8i B = {0, 0, 0, 0, 0, 0, 0, 0};
    B[0] = (int)sel;
    B[2] = (int)sel;
    v8i Cz = {0, 0, 0, 0, 0, 0, 0, 0};
    v8i D = __builtin_amdgcn_wmma_i32_16x16x64_iu8(false, A, false, B, Cz,
                                                   false, false);
    int s = D[0] + D[1] + D[2] + D[3] + D[4] + D[5] + D[6] + D[7];

    __shared__ unsigned lcnt[3];
    if (threadIdx.x < 3) lcnt[threadIdx.x] = 0u;
    __syncthreads();
    unsigned col = lane & 15u;
    if (col < 3u) atomicAdd(&lcnt[col], (unsigned)s);
    __syncthreads();
    if (threadIdx.x < 3) atomicAdd(&gcnt[threadIdx.x], lcnt[threadIdx.x]);
}

__global__ void tsd_finalize(const unsigned* __restrict__ gcnt,
                             float* __restrict__ out, long long n) {
    if (threadIdx.x != 0 || blockIdx.x != 0) return;
    const float P = 1e-7f;
    float cardP = P * (float)n;
    float sum = 0.0f;
    for (int c = 0; c < 3; ++c) {
        unsigned k = gcnt[c];
        float cf    = (float)k;
        float inter = P * cf;
        float card  = cardP + cf;
        float dice  = (2.0f * inter + 1.0f) / ((card + 1.0f) + 1e-7f);
        float lc    = 1.0f - dice;
        if (k > 0u) sum += lc;
    }
    float L = sum / 3.0f;
    out[0] = L + L;
    out[1] = L;
    out[2] = L;
}

extern "C" void kernel_launch(void* const* d_in, const int* in_sizes, int n_in,
                              void* d_out, int out_size, void* d_ws, size_t ws_size,
                              hipStream_t stream) {
    (void)n_in; (void)out_size; (void)ws_size;
    const int* targets = (const int*)d_in[1];
    long long n  = (long long)in_sizes[1];
    long long n4 = n >> 2;

    unsigned* gcnt = (unsigned*)d_ws;
    float* out = (float*)d_out;

    tsd_init_counters<<<1, 32, 0, stream>>>(gcnt);

    const int tpb = 256;
    const long long per = (long long)tpb * 8;
    int blocks = (int)((n4 + per - 1) / per);
    if (blocks < 1) blocks = 1;
    tsd_class_hist_wmma<<<blocks, tpb, 0, stream>>>(
        (const uint4*)targets, n4, targets, n, gcnt);

    tsd_finalize<<<1, 32, 0, stream>>>(gcnt, out, n);
}
